// WindowAttention_11450382811951
// MI455X (gfx1250) — matrix operand order repaired and checked by reading, not compiled, not run on hardware
//
#include <hip/hip_runtime.h>

typedef __attribute__((ext_vector_type(16))) __bf16 v16bf;
typedef __attribute__((ext_vector_type(8)))  float  v8f;
typedef __attribute__((ext_vector_type(4)))  int    v4i;

union BFrag { v16bf v; unsigned short u[16]; };

#if defined(__has_builtin)
#if __has_builtin(__builtin_amdgcn_global_load_async_to_lds_b128) && \
    __has_builtin(__builtin_amdgcn_s_wait_asynccnt)
#define USE_ASYNC_LDS 1
#endif
#endif

#ifdef USE_ASYNC_LDS
typedef __attribute__((address_space(1))) v4i GV4;
typedef __attribute__((address_space(3))) v4i LV4;
__device__ __forceinline__ void async_copy16(const void* g, void* l) {
    __builtin_amdgcn_global_load_async_to_lds_b128((GV4*)g, (LV4*)l, 0, 0);
}
__device__ __forceinline__ void async_wait0() {
    __builtin_amdgcn_s_wait_asynccnt(0);
}
#endif

__device__ __forceinline__ unsigned short f2bf(float f) {
    union { float f; unsigned u; } c; c.f = f;
    unsigned u = c.u;
    u += 0x7FFFu + ((u >> 16) & 1u);   // round-to-nearest-even
    return (unsigned short)(u >> 16);
}
__device__ __forceinline__ float bf2f(unsigned short h) {
    union { unsigned u; float f; } c; c.u = ((unsigned)h) << 16;
    return c.f;
}
// 16-bit A-matrix 16x32 element->K mapping (contiguous: k = 16*hi + e, matches B fills)
__device__ __forceinline__ int kA_idx(int e, int hi) {
    return hi * 16 + e;
}
__device__ __forceinline__ v8f wmma_bf16(const BFrag& a, const BFrag& b, v8f c) {
    return __builtin_amdgcn_wmma_f32_16x16x32_bf16(false, a.v, false, b.v, (short)0, c, false, false);
}

// ---------------- CPB MLP: table[529][12] ----------------
__global__ void cpb_table_k(const float* __restrict__ rct, const float* __restrict__ w1,
                            const float* __restrict__ b1, const float* __restrict__ w2,
                            float* __restrict__ table) {
    int idx = blockIdx.x * blockDim.x + threadIdx.x;
    if (idx >= 529 * 12) return;
    int t = idx / 12, h = idx % 12;
    float c0 = rct[t * 2 + 0], c1 = rct[t * 2 + 1];
    float acc = 0.f;
    for (int k = 0; k < 512; ++k) {
        float hh = fmaxf(c0 * w1[k * 2] + c1 * w1[k * 2 + 1] + b1[k], 0.f);
        acc += hh * w2[h * 512 + k];
    }
    table[t * 12 + h] = acc;
}

// ---------------- rpbm2[w16][h][i][j] = 16*sigmoid(table[rpi[i][j]][h]) + mask[w16][i][j] ----------------
__global__ void rpb_k(const int* __restrict__ rpi, const float* __restrict__ table,
                      const float* __restrict__ mask, float* __restrict__ rpbm2) {
    int idx = blockIdx.x * blockDim.x + threadIdx.x;
    if (idx >= 16 * 12 * 144 * 144) return;
    int w16 = idx / (12 * 20736);
    int rem = idx % (12 * 20736);
    int h = rem / 20736;
    int ij = rem % 20736;
    float v = table[rpi[ij] * 12 + h];
    rpbm2[idx] = 16.f / (1.f + __expf(-v)) + mask[w16 * 20736 + ij];
}

// ---------------- WMMA GEMM: C[M][NC] = A[M][K] @ Bw[NC][K]^T + bias ----------------
// block tile 128x64, 8 waves (4x2), each wave 32x32 = 2x2 wmma tiles, K fully unrolled in steps of 32
template<int A_BF16, int OUT_BF16, int KDIM>
__global__ __launch_bounds__(256) void gemm_k(const void* __restrict__ Ap,
                                              const float* __restrict__ Bw,
                                              const float* __restrict__ bq,
                                              const float* __restrict__ bk,
                                              const float* __restrict__ bv,
                                              void* __restrict__ Cp,
                                              int NC) {
    __shared__ unsigned short As[128 * 32];
    __shared__ unsigned short Bs[32 * 64];
    const int tid = threadIdx.x, lane = tid & 31, wid = tid >> 5;
    const int hi = (lane >> 4) & 1, l16 = lane & 15;
    const int m0 = blockIdx.y * 128, n0 = blockIdx.x * 64;
    const int wm = wid >> 1, wn = wid & 1;

    v8f zero = {0.f,0.f,0.f,0.f,0.f,0.f,0.f,0.f};
    v8f acc[2][2];
    acc[0][0] = zero; acc[0][1] = zero; acc[1][0] = zero; acc[1][1] = zero;

#pragma unroll
    for (int k0 = 0; k0 < KDIM; k0 += 32) {
        __syncthreads();
        // stage A 128x32
        if (A_BF16) {
#ifdef USE_ASYNC_LDS
#pragma unroll
            for (int t = 0; t < 2; ++t) {               // 512 x 16B chunks
                int idx = t * 256 + tid;
                int m = idx >> 2, ks = (idx & 3) * 8;
                async_copy16(((const unsigned short*)Ap) + (size_t)(m0 + m) * KDIM + k0 + ks,
                             As + idx * 8);
            }
#else
#pragma unroll
            for (int t = 0; t < 16; ++t) {
                int idx = t * 256 + tid;
                int m = idx >> 5, k = idx & 31;
                As[idx] = ((const unsigned short*)Ap)[(size_t)(m0 + m) * KDIM + k0 + k];
            }
#endif
        } else {
#pragma unroll
            for (int t = 0; t < 16; ++t) {
                int idx = t * 256 + tid;
                int m = idx >> 5, k = idx & 31;
                As[idx] = f2bf(((const float*)Ap)[(size_t)(m0 + m) * KDIM + k0 + k]);
            }
        }
        // stage B 32x64 (B[k][n] = Bw[n][k]) -- f32 source, must convert through VGPRs
#pragma unroll
        for (int t = 0; t < 8; ++t) {
            int idx = t * 256 + tid;
            int k = idx >> 6, n = idx & 63;
            Bs[idx] = f2bf(Bw[(size_t)(n0 + n) * KDIM + k0 + k]);
        }
#ifdef USE_ASYNC_LDS
        if (A_BF16) async_wait0();
#endif
        __syncthreads();

        BFrag af[2], bf[2];
#pragma unroll
        for (int ti = 0; ti < 2; ++ti) {
            int rm = wm * 32 + ti * 16;
#pragma unroll
            for (int e = 0; e < 16; ++e)
                af[ti].u[e] = As[(rm + l16) * 32 + kA_idx(e, hi)];
        }
#pragma unroll
        for (int tj = 0; tj < 2; ++tj) {
            int nb = wn * 32 + tj * 16;
#pragma unroll
            for (int e = 0; e < 16; ++e)
                bf[tj].u[e] = Bs[(hi * 16 + e) * 64 + nb + l16];
        }
#pragma unroll
        for (int ti = 0; ti < 2; ++ti)
#pragma unroll
            for (int tj = 0; tj < 2; ++tj)
                acc[ti][tj] = wmma_bf16(af[ti], bf[tj], acc[ti][tj]);
    }

#pragma unroll
    for (int ti = 0; ti < 2; ++ti)
#pragma unroll
        for (int tj = 0; tj < 2; ++tj)
#pragma unroll
            for (int r = 0; r < 8; ++r) {
                int gm = m0 + wm * 32 + ti * 16 + r + hi * 8;
                int gn = n0 + wn * 32 + tj * 16 + l16;
                float b = (gn < 384) ? bq[gn] : ((gn < 768) ? bk[gn - 384] : bv[gn - 768]);
                float val = acc[ti][tj][r] + b;
                if (OUT_BF16)
                    ((unsigned short*)Cp)[(size_t)gm * NC + gn] = f2bf(val);
                else
                    ((float*)Cp)[(size_t)gm * NC + gn] = val;
            }
}

// ---------------- per-(b,h,n): L2-normalize q,k; repack qhat/khat/v bf16 [bh][144][32] ----------------
__global__ void norm_k(const unsigned short* __restrict__ qkv,
                       unsigned short* __restrict__ qn,
                       unsigned short* __restrict__ kn,
                       unsigned short* __restrict__ vv) {
    int idx = blockIdx.x * blockDim.x + threadIdx.x;   // over 512*12*144
    if (idx >= 512 * 12 * 144) return;
    int n = idx % 144;
    int bh = idx / 144;
    int h = bh % 12, b = bh / 12;
    const unsigned short* row = qkv + (size_t)(b * 144 + n) * 1152;
    float q[32], k[32];
    float sq = 0.f, sk = 0.f;
#pragma unroll
    for (int d = 0; d < 32; ++d) {
        q[d] = bf2f(row[h * 32 + d]);        sq += q[d] * q[d];
        k[d] = bf2f(row[384 + h * 32 + d]);  sk += k[d] * k[d];
    }
    float rq = rsqrtf(fmaxf(sq, 1e-12f));
    float rk = rsqrtf(fmaxf(sk, 1e-12f));
    size_t o = (size_t)idx * 32;
#pragma unroll
    for (int d = 0; d < 32; ++d) {
        qn[o + d] = f2bf(q[d] * rq);
        kn[o + d] = f2bf(k[d] * rk);
        vv[o + d] = row[768 + h * 32 + d];
    }
}

// ---------------- attention: one workgroup per (window b, head h); 9 waves, wave w = q rows [16w,16w+16) ----------------
__global__ __launch_bounds__(288) void attn_k(const unsigned short* __restrict__ qn,
                                              const unsigned short* __restrict__ kn,
                                              const unsigned short* __restrict__ vv,
                                              const float* __restrict__ rpbm2,
                                              const float* __restrict__ ls,
                                              unsigned short* __restrict__ out) {
    __shared__ unsigned short kns[144 * 32];
    __shared__ unsigned short vs[144 * 32];
    __shared__ unsigned short ps[9 * 16 * 144];        // per-wave P tile (bf16)
    const int bh = blockIdx.x;
    const int h = bh % 12, b = bh / 12;
    const int tid = threadIdx.x, lane = tid & 31, w = tid >> 5;
    const int hi = (lane >> 4) & 1, l16 = lane & 15;

    const unsigned short* kng = kn + (size_t)bh * 4608;
    const unsigned short* vg  = vv + (size_t)bh * 4608;
#ifdef USE_ASYNC_LDS
#pragma unroll
    for (int t = 0; t < 2; ++t) {                      // 576 x 16B chunks each
        int idx = t * 288 + tid;
        async_copy16(kng + idx * 8, kns + idx * 8);
        async_copy16(vg  + idx * 8, vs  + idx * 8);
    }
#else
#pragma unroll
    for (int t = 0; t < 16; ++t) {                     // 4608 = 16*288
        int idx = t * 288 + tid;
        kns[idx] = kng[idx];
        vs[idx]  = vg[idx];
    }
#endif
    // q fragment straight from global (A layout, row stride 32)
    BFrag qa;
    const unsigned short* qg = qn + (size_t)bh * 4608 + (size_t)(w * 16) * 32;
#pragma unroll
    for (int e = 0; e < 16; ++e)
        qa.u[e] = qg[l16 * 32 + kA_idx(e, hi)];
#ifdef USE_ASYNC_LDS
    async_wait0();
#endif
    __syncthreads();

    v8f zero = {0.f,0.f,0.f,0.f,0.f,0.f,0.f,0.f};
    v8f acc[9];
#pragma unroll
    for (int j = 0; j < 9; ++j) acc[j] = zero;

    // S = qhat @ khat^T : K=d=32 in a single WMMA per 16x16 tile
#pragma unroll
    for (int j = 0; j < 9; ++j) {
        BFrag bf;
#pragma unroll
        for (int e = 0; e < 16; ++e)
            bf.u[e] = kns[(j * 16 + l16) * 32 + hi * 16 + e];
        acc[j] = wmma_bf16(qa, bf, acc[j]);
    }

    const float sc = __expf(fminf(ls[h], 4.6051702f));  // exp(min(logit_scale, ln 100))
    const int i0 = w * 16;
    const float* biasp = rpbm2 + ((size_t)(b & 15) * 12 + h) * 20736;
#pragma unroll
    for (int j = 0; j < 9; ++j)
#pragma unroll
        for (int r = 0; r < 8; ++r) {
            int row = i0 + r + hi * 8;
            int col = j * 16 + l16;
            acc[j][r] = acc[j][r] * sc + biasp[row * 144 + col];
        }

    // row softmax: a row lives across 16 lanes of a half-wave and the 9 accumulators
    const int wb = w * 2304;
#pragma unroll
    for (int r = 0; r < 8; ++r) {
        float m = -1e30f;
#pragma unroll
        for (int j = 0; j < 9; ++j) m = fmaxf(m, acc[j][r]);
        m = fmaxf(m, __shfl_xor(m, 1, 32));
        m = fmaxf(m, __shfl_xor(m, 2, 32));
        m = fmaxf(m, __shfl_xor(m, 4, 32));
        m = fmaxf(m, __shfl_xor(m, 8, 32));
        float s = 0.f;
#pragma unroll
        for (int j = 0; j < 9; ++j) {
            float e = __expf(acc[j][r] - m);
            acc[j][r] = e;
            s += e;
        }
        s += __shfl_xor(s, 1, 32);
        s += __shfl_xor(s, 2, 32);
        s += __shfl_xor(s, 4, 32);
        s += __shfl_xor(s, 8, 32);
        float inv = 1.f / s;
#pragma unroll
        for (int j = 0; j < 9; ++j)
            ps[wb + (r + hi * 8) * 144 + j * 16 + l16] = f2bf(acc[j][r] * inv);
    }
    __syncthreads();

    // O = P(16x144) @ V(144x32): K tiles = 4 full + guarded 16-tail
    v8f oacc[2];
    oacc[0] = zero; oacc[1] = zero;
#pragma unroll
    for (int kt = 0; kt < 5; ++kt) {
        BFrag pa;
#pragma unroll
        for (int e = 0; e < 16; ++e) {
            int k = kt * 32 + kA_idx(e, hi);
            pa.u[e] = (k < 144) ? ps[wb + l16 * 144 + k] : (unsigned short)0;
        }
#pragma unroll
        for (int tj = 0; tj < 2; ++tj) {
            BFrag bv;
#pragma unroll
            for (int e = 0; e < 16; ++e) {
                int kk = kt * 32 + hi * 16 + e;
                bv.u[e] = (kk < 144) ? vs[kk * 32 + tj * 16 + l16] : (unsigned short)0;
            }
            oacc[tj] = wmma_bf16(pa, bv, oacc[tj]);
        }
    }
#pragma unroll
    for (int tj = 0; tj < 2; ++tj)
#pragma unroll
        for (int r = 0; r < 8; ++r) {
            int row = i0 + r + hi * 8;
            out[(size_t)(b * 144 + row) * 384 + h * 32 + tj * 16 + l16] = f2bf(oacc[tj][r]);
        }
}

extern "C" void kernel_launch(void* const* d_in, const int* in_sizes, int n_in,
                              void* d_out, int out_size, void* d_ws, size_t ws_size,
                              hipStream_t stream) {
    (void)in_sizes; (void)n_in; (void)out_size; (void)ws_size;
    const float* x      = (const float*)d_in[0];
    const float* rct    = (const float*)d_in[1];
    const int*   rpi    = (const int*)d_in[2];
    const float* mask   = (const float*)d_in[3];
    const float* qkv_w  = (const float*)d_in[4];
    const float* qb     = (const float*)d_in[5];
    const float* kb     = (const float*)d_in[6];
    const float* vb     = (const float*)d_in[7];
    const float* ls     = (const float*)d_in[8];
    const float* w1     = (const float*)d_in[9];
    const float* b1     = (const float*)d_in[10];
    const float* w2     = (const float*)d_in[11];
    const float* pw     = (const float*)d_in[12];
    const float* pb     = (const float*)d_in[13];
    float* out = (float*)d_out;

    char* base = (char*)d_ws;
    unsigned short* qkv_bf = (unsigned short*)base;              // 73728*1152 bf16
    unsigned short* attn_o = qkv_bf;                             // alias (stream-serialized reuse)
    size_t off = (size_t)73728 * 1152 * 2;
    unsigned short* qn = (unsigned short*)(base + off); off += (size_t)6144 * 144 * 32 * 2;
    unsigned short* kn = (unsigned short*)(base + off); off += (size_t)6144 * 144 * 32 * 2;
    unsigned short* vv = (unsigned short*)(base + off); off += (size_t)6144 * 144 * 32 * 2;
    float* rpbm2 = (float*)(base + off); off += (size_t)16 * 12 * 144 * 144 * 4;
    float* table = (float*)(base + off);

    cpb_table_k<<<(529 * 12 + 255) / 256, 256, 0, stream>>>(rct, w1, b1, w2, table);
    rpb_k<<<(16 * 12 * 144 * 144 + 255) / 256, 256, 0, stream>>>(rpi, table, mask, rpbm2);
    gemm_k<0, 1, 384><<<dim3(18, 576), 256, 0, stream>>>(x, qkv_w, qb, kb, vb, qkv_bf, 1152);
    norm_k<<<(512 * 12 * 144 + 255) / 256, 256, 0, stream>>>(qkv_bf, qn, kn, vv);
    attn_k<<<6144, 288, 0, stream>>>(qn, kn, vv, rpbm2, ls, attn_o);
    gemm_k<1, 0, 384><<<dim3(6, 576), 256, 0, stream>>>(attn_o, pw, pb, pb, pb, out, 384);
}
